// VQ_VAE_73624329388439
// MI455X (gfx1250) — matrix operand order repaired and checked by reading, not compiled, not run on hardware
//
#include <hip/hip_runtime.h>
#include <hip/hip_bf16.h>
#include <math.h>
#include <stdint.h>

typedef __attribute__((ext_vector_type(16))) _Float16 v16h;
typedef __attribute__((ext_vector_type(8)))  float    v8f;

#define EPS 1e-5f

// ---------------------------------------------------------------------------
// init: zero stats + loss accumulator region
// ---------------------------------------------------------------------------
__global__ void k_init(float* __restrict__ p, int n) {
    int i = blockIdx.x * blockDim.x + threadIdx.x;
    if (i < n) p[i] = 0.0f;
}

// ---------------------------------------------------------------------------
// conv1: direct f32, Cin=3 (K=48 too small for WMMA). in [N,3,256,256] ->
// raw out [N,16,128,128]
// ---------------------------------------------------------------------------
__global__ void k_conv1(const float* __restrict__ x, const float* __restrict__ w,
                        float* __restrict__ out) {
    const int HW = 128 * 128;
    size_t gid = (size_t)blockIdx.x * blockDim.x + threadIdx.x;
    size_t total = (size_t)64 * 16 * HW;
    if (gid >= total) return;
    int ox = gid % 128;
    int oy = (gid / 128) % 128;
    int co = (gid / HW) % 16;
    int n  = gid / ((size_t)HW * 16);
    float acc = 0.0f;
    #pragma unroll
    for (int ci = 0; ci < 3; ++ci) {
        #pragma unroll
        for (int ky = 0; ky < 4; ++ky) {
            int iy = 2 * oy - 1 + ky;
            if (iy < 0 || iy >= 256) continue;
            #pragma unroll
            for (int kx = 0; kx < 4; ++kx) {
                int ix = 2 * ox - 1 + kx;
                if (ix < 0 || ix >= 256) continue;
                acc += x[(((size_t)n * 3 + ci) * 256 + iy) * 256 + ix]
                     * w[((co * 3 + ci) * 4 + ky) * 4 + kx];
            }
        }
    }
    out[gid] = acc;
}

// ---------------------------------------------------------------------------
// Implicit-GEMM WMMA conv: k=4 s=2 p=1. One wave computes 16 pixels x COT
// out-channels (COT/16 accumulators share the A fragment). Input BN+ReLU of
// the previous layer is fused into patch staging (fuse=1).
// K ordering: k = ci*16 + (ky*4+kx)  ->  A K-pairs are LDS-adjacent (b32).
// Weights staged as packed half-pairs [k/2][COT] (b32 B-fragment loads).
// grid = (Wout/16, N*Hout, Cout/COT), block = 32.
// ---------------------------------------------------------------------------
template<int CIN, int COT>
__global__ __launch_bounds__(32)
void k_conv_wmma(const float* __restrict__ in, const float* __restrict__ w,
                 float* __restrict__ out,
                 const float* __restrict__ bns, const float* __restrict__ bnb,
                 int fuse, int N, int Cout, int Hin, int Win) {
    const int Hout = Hin >> 1, Wout = Win >> 1;
    const int KT = CIN * 16;

    const int ox0 = blockIdx.x * 16;
    const int oy  = blockIdx.y % Hout;
    const int n   = blockIdx.y / Hout;
    const int co0 = blockIdx.z * COT;
    const int lane = threadIdx.x;
    const int m  = lane & 15;
    const int hs = lane >> 4;

    __shared__ __align__(16) _Float16 sP[CIN * 4 * 34]; // [ci][ry 0..3][cx 0..33]
    __shared__ __align__(16) unsigned sWp[(CIN * 8) * COT]; // packed [k/2][co]

    // stage input patch (f32 -> f16), optional fused BN+ReLU
    for (int t = lane; t < CIN * 4 * 34; t += 32) {
        int cx = t % 34;
        int ry = (t / 34) & 3;
        int ci = t / (34 * 4);
        int iy = 2 * oy - 1 + ry;
        int ix = 2 * ox0 - 1 + cx;
        float v = 0.0f;
        if (iy >= 0 && iy < Hin && ix >= 0 && ix < Win) {
            v = in[(((size_t)n * CIN + ci) * Hin + iy) * Win + ix];
            if (fuse) v = fmaxf(v * bns[ci] + bnb[ci], 0.0f);
        }
        sP[t] = (_Float16)v;
    }
    // stage weight tile: k = ci*16+kk is exactly the OIHW flat minor index
    for (int t = lane; t < (KT / 2) * COT; t += 32) {
        int co = t % COT;
        int k2 = t / COT;
        const float* wp = w + (size_t)(co0 + co) * KT + 2 * k2;
        union { _Float16 h[2]; unsigned u; } pk;
        pk.h[0] = (_Float16)wp[0];
        pk.h[1] = (_Float16)wp[1];
        sWp[t] = pk.u;
    }
    __syncthreads();

    const unsigned* sP32 = (const unsigned*)sP;
    v8f acc[COT / 16];
    #pragma unroll
    for (int ct = 0; ct < COT / 16; ++ct) acc[ct] = (v8f){};

    for (int kb = 0; kb < KT; kb += 32) {
        union { v16h v; unsigned u[8]; } A;
        #pragma unroll
        for (int r = 0; r < 8; ++r) {
            int ka = kb + ((r >> 2) * 16 + hs * 8 + ((r & 3) << 1));
            int ci = ka >> 4, kk = ka & 15, ky = kk >> 2, kx = kk & 3;
            // pair (ka, ka+1) = (kx, kx+1): adjacent halves, aligned b32
            A.u[r] = sP32[(ci * 4 + ky) * 17 + m + (kx >> 1)];
        }
        #pragma unroll
        for (int ct = 0; ct < COT / 16; ++ct) {
            union { v16h v; unsigned u[8]; } B;
            #pragma unroll
            for (int r = 0; r < 8; ++r)
                B.u[r] = sWp[((kb >> 1) + (r >> 2) * 8 + hs * 4 + (r & 3)) * COT + ct * 16 + m];
            acc[ct] = __builtin_amdgcn_wmma_f32_16x16x32_f16(
                        false, A.v, false, B.v, (short)0, acc[ct], false, false);
        }
    }

    #pragma unroll
    for (int ct = 0; ct < COT / 16; ++ct) {
        int co = co0 + ct * 16 + m;
        size_t base = (((size_t)n * Cout + co) * Hout + oy) * Wout + ox0;
        #pragma unroll
        for (int r = 0; r < 8; ++r)
            out[base + r + 8 * hs] = acc[ct][r];
    }
}

// ---------------------------------------------------------------------------
// WMMA transposed conv (k=4 s=2 p=1) by output-parity decomposition: class
// (py,px) is a 2x2 stride-1 conv, K = 4*CIN, k = ci*4 + (dy*2+dx).
// grid = (Win/16, N*Hin, 4*Cout/16), block = 32.
// ---------------------------------------------------------------------------
template<int CIN>
__global__ __launch_bounds__(32)
void k_deconv_wmma(const float* __restrict__ in, const float* __restrict__ w,
                   float* __restrict__ out,
                   const float* __restrict__ bns, const float* __restrict__ bnb,
                   int fuse, int N, int Cout, int Hin, int Win) {
    const int KT = CIN * 4;
    const int x0 = blockIdx.x * 16;
    const int y  = blockIdx.y % Hin;
    const int n  = blockIdx.y / Hin;
    const int coT = Cout >> 4;
    const int cls = blockIdx.z / coT;      // 0..3 : (py,px)
    const int co0 = (blockIdx.z % coT) * 16;
    const int py = cls >> 1, px = cls & 1;
    const int lane = threadIdx.x;
    const int m  = lane & 15;
    const int hs = lane >> 4;

    __shared__ __align__(16) _Float16 sP[CIN * 2 * 17];  // [ci][dy 0..1][c 0..16]
    __shared__ __align__(16) unsigned sWp[(CIN * 2) * 16]; // packed [k/2][co]

    for (int t = lane; t < CIN * 2 * 17; t += 32) {
        int c  = t % 17;
        int dy = (t / 17) & 1;
        int ci = t / 34;
        int iy = y + py - 1 + dy;
        int ix = x0 + px - 1 + c;
        float v = 0.0f;
        if (iy >= 0 && iy < Hin && ix >= 0 && ix < Win) {
            v = in[(((size_t)n * CIN + ci) * Hin + iy) * Win + ix];
            if (fuse) v = fmaxf(v * bns[ci] + bnb[ci], 0.0f);
        }
        sP[t] = (_Float16)v;
    }
    // weights: k = ci*4 + tt, tt = dy*2+dx ; value w[co][ci][py+2dy][px+2dx]
    for (int t = lane; t < (KT / 2) * 16; t += 32) {
        int co = t & 15;
        int k2 = t >> 4;
        union { _Float16 h[2]; unsigned u; } pk;
        #pragma unroll
        for (int e = 0; e < 2; ++e) {
            int k  = 2 * k2 + e;
            int ci = k >> 2, tt = k & 3, dy = tt >> 1, dx = tt & 1;
            pk.h[e] = (_Float16)w[((size_t)(co0 + co) * CIN + ci) * 16
                                  + (py + 2 * dy) * 4 + (px + 2 * dx)];
        }
        sWp[t] = pk.u;
    }
    __syncthreads();

    v8f acc = {};
    for (int kb = 0; kb < KT; kb += 32) {
        union { v16h v; _Float16 h[16]; } A;
        union { v16h v; unsigned u[8]; } B;
        #pragma unroll
        for (int r = 0; r < 8; ++r) {
            int ka = kb + ((r >> 2) * 16 + hs * 8 + ((r & 3) << 1));
            int ci = ka >> 2, tt = ka & 3, dy = tt >> 1;
            // pair (ka, ka+1): (dx=0, dx=1), same ci/dy
            int base = (ci * 2 + dy) * 17 + m;
            A.h[2 * r]     = sP[base];
            A.h[2 * r + 1] = sP[base + 1];
            B.u[r] = sWp[((kb >> 1) + (r >> 2) * 8 + hs * 4 + (r & 3)) * 16 + m];
        }
        acc = __builtin_amdgcn_wmma_f32_16x16x32_f16(
                false, A.v, false, B.v, (short)0, acc, false, false);
    }

    const int co = co0 + m;
    const int Wout = Win * 2;
    size_t rowbase = (((size_t)n * Cout + co) * (2 * Hin) + 2 * y + py)
                     * (size_t)Wout + px;
    #pragma unroll
    for (int r = 0; r < 8; ++r) {
        int mm = r + 8 * hs;
        out[rowbase + 2 * (size_t)(x0 + mm)] = acc[r];
    }
}

// ---------------------------------------------------------------------------
// BatchNorm (training) statistics over raw conv output
// ---------------------------------------------------------------------------
#define BPC 64
__global__ void k_bn_stats(const float* __restrict__ x, float* __restrict__ sum,
                           float* __restrict__ sumsq, int C, int HW) {
    int c = blockIdx.x / BPC;
    int b = blockIdx.x % BPC;
    size_t per = (size_t)64 * HW;
    float s = 0.0f, s2 = 0.0f;
    for (size_t j = (size_t)b * 256 + threadIdx.x; j < per; j += (size_t)BPC * 256) {
        int n  = j / HW;
        int hw = j % HW;
        float v = x[((size_t)n * C + c) * HW + hw];
        s += v; s2 += v * v;
    }
    __shared__ float r1[256], r2[256];
    r1[threadIdx.x] = s; r2[threadIdx.x] = s2;
    __syncthreads();
    for (int st = 128; st > 0; st >>= 1) {
        if ((int)threadIdx.x < st) {
            r1[threadIdx.x] += r1[threadIdx.x + st];
            r2[threadIdx.x] += r2[threadIdx.x + st];
        }
        __syncthreads();
    }
    if (threadIdx.x == 0) {
        atomicAdd(&sum[c], r1[0]);
        atomicAdd(&sumsq[c], r2[0]);
    }
}

__global__ void k_bn_finalize(const float* __restrict__ sum, const float* __restrict__ sumsq,
                              const float* __restrict__ g, const float* __restrict__ bt,
                              float* __restrict__ scale, float* __restrict__ shift,
                              int C, float invcnt) {
    int c = threadIdx.x;
    if (c >= C) return;
    float mean = sum[c] * invcnt;
    float var  = sumsq[c] * invcnt - mean * mean;
    float sc   = g[c] * rsqrtf(var + EPS);
    scale[c] = sc;
    shift[c] = bt[c] - mean * sc;
}

// last layer only: BN + tanh into d_out
__global__ void k_bn_apply_tanh(const float* __restrict__ in, float* __restrict__ out,
                                const float* __restrict__ scale, const float* __restrict__ shift,
                                int C, int HW, size_t total) {
    size_t gid = (size_t)blockIdx.x * blockDim.x + threadIdx.x;
    if (gid >= total) return;
    int c = (int)((gid / HW) % C);
    out[gid] = tanhf(in[gid] * scale[c] + shift[c]);
}

// ---------------------------------------------------------------------------
// VQ prep: raw z_e conv output + fused BN+ReLU -> zh [65536][64] f16 + znorm
// ---------------------------------------------------------------------------
__global__ void k_vq_prep_z(const float* __restrict__ raw,
                            const float* __restrict__ scale, const float* __restrict__ shift,
                            _Float16* __restrict__ zh, float* __restrict__ zn) {
    int i = blockIdx.x;          // 0..65535
    int d = threadIdx.x;         // 0..63  (== channel)
    int n = i >> 10, hw = i & 1023;
    float v = fmaxf(raw[((size_t)n * 64 + d) * 1024 + hw] * scale[d] + shift[d], 0.0f);
    zh[(size_t)i * 64 + d] = (_Float16)v;
    __shared__ float red[64];
    red[d] = v * v;
    __syncthreads();
    if (d < 32) red[d] += red[d + 32];
    __syncthreads();
    if (d == 0) {
        float s = 0.0f;
        for (int t = 0; t < 32; ++t) s += red[t];
        zn[i] = s;
    }
}

// codebook [512,64] -> packed half-pair transpose cbtp [32 pairs][512] + cnorm
__global__ void k_vq_prep_cb(const float* __restrict__ cb, unsigned* __restrict__ cbtp,
                             float* __restrict__ cn) {
    int k = blockIdx.x;          // 0..511
    int d = threadIdx.x;         // 0..63
    float v = cb[(size_t)k * 64 + d];
    __shared__ _Float16 sh[64];
    __shared__ float red[64];
    sh[d] = (_Float16)v;
    red[d] = v * v;
    __syncthreads();
    if (d < 32) {
        union { _Float16 h[2]; unsigned u; } pk;
        pk.h[0] = sh[2 * d];
        pk.h[1] = sh[2 * d + 1];
        cbtp[(size_t)d * 512 + k] = pk.u;
        red[d] += red[d + 32];
    }
    __syncthreads();
    if (d == 0) {
        float s = 0.0f;
        for (int t = 0; t < 32; ++t) s += red[t];
        cn[k] = s;
    }
}

// ---------------------------------------------------------------------------
// VQ argmin: one wave per 16 z rows x all 512 codes. z tile staged to LDS via
// CDNA5 async global->LDS (ASYNCcnt); B fragments are packed b32 loads.
// ---------------------------------------------------------------------------
__global__ __launch_bounds__(32)
void k_vq_argmin(const _Float16* __restrict__ zh, const unsigned* __restrict__ cbtp,
                 const float* __restrict__ zn, const float* __restrict__ cn,
                 int* __restrict__ idx) {
    const int i0 = blockIdx.x * 16;
    const int lane = threadIdx.x;
    const int m  = lane & 15;
    const int hs = lane >> 4;

    __shared__ __align__(16) _Float16 sZ[16 * 64];   // sole LDS object -> offset 0

    // async copy 2048B of zh into sZ: 32 lanes x 16B, x4 with matching
    // LDS/global instruction offsets (offset applies to both sides).
    {
        unsigned lds_addr = (unsigned)lane * 16u;    // sZ starts at LDS offset 0
        uint64_t ga = (uint64_t)(uintptr_t)(zh + (size_t)i0 * 64) + (unsigned)lane * 16u;
        asm volatile(
            "global_load_async_to_lds_b128 %0, %1, off\n\t"
            "global_load_async_to_lds_b128 %0, %1, off offset:512\n\t"
            "global_load_async_to_lds_b128 %0, %1, off offset:1024\n\t"
            "global_load_async_to_lds_b128 %0, %1, off offset:1536\n\t"
            "s_wait_asynccnt 0x0"
            :
            : "v"(lds_addr), "v"(ga)
            : "memory");
    }
    __syncthreads();

    // A fragments: N-tile invariant, built once (aligned b32 LDS reads)
    const unsigned* sZ32 = (const unsigned*)sZ;
    union { v16h v; unsigned u[8]; } A0, A1;
    #pragma unroll
    for (int r = 0; r < 8; ++r) {
        int ka = (r >> 2) * 16 + hs * 8 + ((r & 3) << 1);
        A0.u[r] = sZ32[m * 32 + (ka >> 1)];
        A1.u[r] = sZ32[m * 32 + ((32 + ka) >> 1)];
    }

    float znl[8];
    #pragma unroll
    for (int r = 0; r < 8; ++r) znl[r] = zn[i0 + r + 8 * hs];

    float bestd[8];
    int   besti[8];
    #pragma unroll
    for (int r = 0; r < 8; ++r) { bestd[r] = 3.4e38f; besti[r] = 0; }

    for (int nt = 0; nt < 32; ++nt) {
        union { v16h v; unsigned u[8]; } B0, B1;
        #pragma unroll
        for (int r = 0; r < 8; ++r) {
            int p = (r >> 2) * 8 + hs * 4 + (r & 3); // K rows (2p, 2p+1)
            B0.u[r] = cbtp[(size_t)(p     ) * 512 + nt * 16 + m];
            B1.u[r] = cbtp[(size_t)(p + 16) * 512 + nt * 16 + m];
        }
        v8f acc = {};
        acc = __builtin_amdgcn_wmma_f32_16x16x32_f16(false, A0.v, false, B0.v,
                                                     (short)0, acc, false, false);
        acc = __builtin_amdgcn_wmma_f32_16x16x32_f16(false, A1.v, false, B1.v,
                                                     (short)0, acc, false, false);
        float cnn = cn[nt * 16 + m];
        #pragma unroll
        for (int r = 0; r < 8; ++r) {
            float d = znl[r] + cnn - 2.0f * acc[r];
            int   ci = nt * 16 + m;
            if (d < bestd[r] || (d == bestd[r] && ci < besti[r])) {
                bestd[r] = d; besti[r] = ci;
            }
        }
    }

    #pragma unroll
    for (int s = 8; s >= 1; s >>= 1) {
        #pragma unroll
        for (int r = 0; r < 8; ++r) {
            float od = __shfl_xor(bestd[r], s, 32);
            int   oi = __shfl_xor(besti[r], s, 32);
            if (od < bestd[r] || (od == bestd[r] && oi < besti[r])) {
                bestd[r] = od; besti[r] = oi;
            }
        }
    }
    if (m == 0) {
        #pragma unroll
        for (int r = 0; r < 8; ++r) idx[i0 + r + 8 * hs] = besti[r];
    }
}

// gather quant (straight-through forward == quant) + loss sum; z from raw+BN
__global__ void k_vq_gather_loss(const float* __restrict__ raw,
                                 const float* __restrict__ scale, const float* __restrict__ shift,
                                 const float* __restrict__ cb, const int* __restrict__ idx,
                                 float* __restrict__ quant, float* __restrict__ lacc) {
    size_t gid = (size_t)blockIdx.x * 256 + threadIdx.x;  // 65536*64 total
    int i = (int)(gid >> 6), d = (int)(gid & 63);
    int n = i >> 10, hw = i & 1023;
    float z = fmaxf(raw[((size_t)n * 64 + d) * 1024 + hw] * scale[d] + shift[d], 0.0f);
    float q = cb[(size_t)idx[i] * 64 + d];
    quant[((size_t)n * 64 + d) * 1024 + hw] = q;
    float df = q - z;
    __shared__ float red[256];
    red[threadIdx.x] = df * df;
    __syncthreads();
    for (int s = 128; s > 0; s >>= 1) {
        if ((int)threadIdx.x < s) red[threadIdx.x] += red[threadIdx.x + s];
        __syncthreads();
    }
    if (threadIdx.x == 0) atomicAdd(lacc, red[0]);
}

__global__ void k_loss_final(const float* __restrict__ lacc, float* __restrict__ out) {
    out[0] = 2.0f * lacc[0] / (65536.0f * 64.0f);   // e_loss + q_loss
}

// ---------------------------------------------------------------------------
// direct deconv (last layer, Cout=3 < one WMMA tile), fused input BN+ReLU
// ---------------------------------------------------------------------------
__global__ void k_deconv_direct(const float* __restrict__ in, const float* __restrict__ w,
                                float* __restrict__ out,
                                const float* __restrict__ bns, const float* __restrict__ bnb,
                                int N, int Cin, int Cout, int Hin, int Win) {
    int Hout = Hin * 2, Wout = Win * 2;
    size_t total = (size_t)N * Cout * Hout * Wout;
    size_t gid = (size_t)blockIdx.x * blockDim.x + threadIdx.x;
    if (gid >= total) return;
    int ox = gid % Wout;
    int oy = (gid / Wout) % Hout;
    int co = (gid / ((size_t)Wout * Hout)) % Cout;
    int n  = gid / ((size_t)Wout * Hout * Cout);
    float acc = 0.0f;
    #pragma unroll
    for (int ky = 0; ky < 4; ++ky) {
        int ty = oy + ky - 2;
        if (ty & 1) continue;
        int iy = ty >> 1;
        if (iy < 0 || iy >= Hin) continue;
        #pragma unroll
        for (int kx = 0; kx < 4; ++kx) {
            int tx = ox + kx - 2;
            if (tx & 1) continue;
            int ix = tx >> 1;
            if (ix < 0 || ix >= Win) continue;
            const float* ip = in + (((size_t)n * Cin) * Hin + iy) * Win + ix;
            const float* wp = w + (size_t)co * Cin * 16 + ky * 4 + kx;
            for (int ci = 0; ci < Cin; ++ci) {
                float v = fmaxf(ip[(size_t)ci * Hin * Win] * bns[ci] + bnb[ci], 0.0f);
                acc += v * wp[(size_t)ci * 16];
            }
        }
    }
    out[gid] = acc;
}

// ---------------------------------------------------------------------------
// host side
// ---------------------------------------------------------------------------
extern "C" void kernel_launch(void* const* d_in, const int* in_sizes, int n_in,
                              void* d_out, int out_size, void* d_ws, size_t ws_size,
                              hipStream_t stream) {
    const float* x   = (const float*)d_in[0];
    const float* cb  = (const float*)d_in[1];
    const float* eW1 = (const float*)d_in[2];
    const float* eg1 = (const float*)d_in[4];
    const float* ebt1= (const float*)d_in[5];
    const float* eW2 = (const float*)d_in[6];
    const float* eg2 = (const float*)d_in[8];
    const float* ebt2= (const float*)d_in[9];
    const float* eW3 = (const float*)d_in[10];
    const float* eg3 = (const float*)d_in[12];
    const float* ebt3= (const float*)d_in[13];
    const float* dW1 = (const float*)d_in[14];
    const float* dg1 = (const float*)d_in[16];
    const float* dbt1= (const float*)d_in[17];
    const float* dW2 = (const float*)d_in[18];
    const float* dg2 = (const float*)d_in[20];
    const float* dbt2= (const float*)d_in[21];
    const float* dW3 = (const float*)d_in[22];
    const float* dg3 = (const float*)d_in[24];
    const float* dbt3= (const float*)d_in[25];
    float* zout = (float*)d_out;
    float* lout = zout + (size_t)64 * 3 * 256 * 256;

    // workspace layout (all block sizes are multiples of 4 floats -> ZH 16B aligned)
    float* STATS = (float*)d_ws;               // 4096 floats
    float* LACC  = STATS + 1600;
    float* R1 = STATS + 4096;                  // 16777216  (conv1 raw / deconv2 raw)
    float* R2 = R1 + 16777216;                 //  8388608  (conv2 raw / deconv1 raw)
    float* R3 = R2 + 8388608;                  //  4194304  (conv3 raw = z_e pre-BN)
    float* R4 = R3 + 4194304;                  //  4194304  (quant)
    float* R5 = R4 + 4194304;                  // 12582912  (deconv3 raw)
    _Float16* ZH = (_Float16*)(R5 + 12582912); // 65536*64 halves (16B aligned)
    unsigned* CBTP = (unsigned*)(ZH + (size_t)65536 * 64); // 32*512 packed pairs
    float* ZN = (float*)(CBTP + 32 * 512);     // 65536
    float* CN = ZN + 65536;                    // 512
    int*   IDX = (int*)(CN + 512);             // 65536

    auto SUM   = [&](int L) { return STATS + L * 256; };
    auto SUMSQ = [&](int L) { return STATS + L * 256 + 64; };
    auto SCALE = [&](int L) { return STATS + L * 256 + 128; };
    auto SHIFT = [&](int L) { return STATS + L * 256 + 192; };

    k_init<<<16, 256, 0, stream>>>(STATS, 4096);

    // encoder conv1 (direct) -> R1 raw; BN0 stats
    {
        size_t total = (size_t)64 * 16 * 128 * 128;
        k_conv1<<<(int)((total + 255) / 256), 256, 0, stream>>>(x, eW1, R1);
        k_bn_stats<<<16 * BPC, 256, 0, stream>>>(R1, SUM(0), SUMSQ(0), 16, 128 * 128);
        k_bn_finalize<<<1, 64, 0, stream>>>(SUM(0), SUMSQ(0), eg1, ebt1, SCALE(0), SHIFT(0),
                                            16, 1.0f / (64.0f * 128 * 128));
    }
    // conv2 (WMMA, fused BN0+ReLU on input) -> R2 raw; BN1 stats
    {
        dim3 g(64 / 16, 64 * 64, 1);
        k_conv_wmma<16, 32><<<g, 32, 0, stream>>>(R1, eW2, R2, SCALE(0), SHIFT(0), 1,
                                                  64, 32, 128, 128);
        k_bn_stats<<<32 * BPC, 256, 0, stream>>>(R2, SUM(1), SUMSQ(1), 32, 64 * 64);
        k_bn_finalize<<<1, 64, 0, stream>>>(SUM(1), SUMSQ(1), eg2, ebt2, SCALE(1), SHIFT(1),
                                            32, 1.0f / (64.0f * 64 * 64));
    }
    // conv3 (WMMA, fused BN1+ReLU) -> R3 raw; BN2 stats
    {
        dim3 g(32 / 16, 64 * 32, 2);
        k_conv_wmma<32, 32><<<g, 32, 0, stream>>>(R2, eW3, R3, SCALE(1), SHIFT(1), 1,
                                                  64, 64, 64, 64);
        k_bn_stats<<<64 * BPC, 256, 0, stream>>>(R3, SUM(2), SUMSQ(2), 64, 32 * 32);
        k_bn_finalize<<<1, 64, 0, stream>>>(SUM(2), SUMSQ(2), eg3, ebt3, SCALE(2), SHIFT(2),
                                            64, 1.0f / (64.0f * 32 * 32));
    }
    // vector quantization (BN2+ReLU fused into prep/gather)
    k_vq_prep_z<<<65536, 64, 0, stream>>>(R3, SCALE(2), SHIFT(2), ZH, ZN);
    k_vq_prep_cb<<<512, 64, 0, stream>>>(cb, CBTP, CN);
    k_vq_argmin<<<65536 / 16, 32, 0, stream>>>(ZH, CBTP, ZN, CN, IDX);
    k_vq_gather_loss<<<(65536 * 64) / 256, 256, 0, stream>>>(R3, SCALE(2), SHIFT(2),
                                                             cb, IDX, R4, LACC);
    k_loss_final<<<1, 1, 0, stream>>>(LACC, lout);

    // deconv1 (WMMA parity, no input BN) 64->32, 32x32->64x64 -> R2 raw; BN3
    {
        dim3 g(32 / 16, 64 * 32, 4 * (32 / 16));
        k_deconv_wmma<64><<<g, 32, 0, stream>>>(R4, dW1, R2, nullptr, nullptr, 0,
                                                64, 32, 32, 32);
        k_bn_stats<<<32 * BPC, 256, 0, stream>>>(R2, SUM(3), SUMSQ(3), 32, 64 * 64);
        k_bn_finalize<<<1, 64, 0, stream>>>(SUM(3), SUMSQ(3), dg1, dbt1, SCALE(3), SHIFT(3),
                                            32, 1.0f / (64.0f * 64 * 64));
    }
    // deconv2 (WMMA parity, fused BN3+ReLU) 32->16, 64x64->128x128 -> R1 raw; BN4
    {
        dim3 g(64 / 16, 64 * 64, 4 * (16 / 16));
        k_deconv_wmma<32><<<g, 32, 0, stream>>>(R2, dW2, R1, SCALE(3), SHIFT(3), 1,
                                                64, 16, 64, 64);
        k_bn_stats<<<16 * BPC, 256, 0, stream>>>(R1, SUM(4), SUMSQ(4), 16, 128 * 128);
        k_bn_finalize<<<1, 64, 0, stream>>>(SUM(4), SUMSQ(4), dg2, dbt2, SCALE(4), SHIFT(4),
                                            16, 1.0f / (64.0f * 128 * 128));
    }
    // deconv3 (direct, fused BN4+ReLU) 16->3 -> R5 raw; BN5 + tanh -> d_out
    {
        size_t total = (size_t)64 * 3 * 256 * 256;
        k_deconv_direct<<<(int)((total + 255) / 256), 256, 0, stream>>>(
            R1, dW3, R5, SCALE(4), SHIFT(4), 64, 16, 3, 128, 128);
        k_bn_stats<<<3 * BPC, 256, 0, stream>>>(R5, SUM(5), SUMSQ(5), 3, 256 * 256);
        k_bn_finalize<<<1, 64, 0, stream>>>(SUM(5), SUMSQ(5), dg3, dbt3, SCALE(5), SHIFT(5),
                                            3, 1.0f / (64.0f * 256 * 256));
        k_bn_apply_tanh<<<(int)((total + 255) / 256), 256, 0, stream>>>(
            R5, zout, SCALE(5), SHIFT(5), 3, 256 * 256, total);
    }
    (void)in_sizes; (void)n_in; (void)out_size; (void)ws_size;
}
